// MoE_Layer_79912161509419
// MI455X (gfx1250) — compile-verified, matrix operand order checked by reading, not run on hardware
//
#include <hip/hip_runtime.h>
#include <hip/hip_bf16.h>
#include <math.h>

// ---------------------------------------------------------------------------
// MoE layer for MI455X (gfx1250), bf16 WMMA path with fragment-packed weights.
//   B=4, S=2048 -> N=8192 tokens, D=1024, H=4096, E=8, top-2 routing.
// ---------------------------------------------------------------------------

#define N_TOK   8192
#define DMODEL  1024
#define HDIM    4096
#define NEXP    8
#define MTILE   32                     // token rows per FFN block (2 WMMA M-subtiles)
#define TILES_M (N_TOK / MTILE)        // 256 worst-case tiles per expert
#define SLAB    512                    // hidden columns per LDS slab
#define XSTR    2096                   // bytes/row, X tile in LDS (2048 + pad, /4 % 64 == 12)
#define HSTR    1072                   // bytes/row, hidden slab in LDS (1024 + pad)

#define KC1 (DMODEL / 32)              // 32  k-chunks, GEMM1
#define NC1 (HDIM / 16)                // 256 n-chunks, GEMM1
#define KC2 (HDIM / 32)                // 128 k-chunks, GEMM2
#define NC2 (DMODEL / 16)              // 64  n-chunks, GEMM2
#define W1T_BYTES_PER_E ((size_t)KC1 * NC1 * 1024)   // 8 MB
#define W2T_BYTES_PER_E ((size_t)KC2 * NC2 * 1024)   // 8 MB

typedef __attribute__((ext_vector_type(16))) __bf16 v16bf;
typedef __attribute__((ext_vector_type(8)))  float  v8f;

__device__ __forceinline__ float gelu_exact(float v) {
    return 0.5f * v * (1.0f + erff(v * 0.7071067811865475f));
}

__device__ __forceinline__ v8f wmma_bf16(v16bf a, v16bf b, v8f c) {
    return __builtin_amdgcn_wmma_f32_16x16x32_bf16(false, a, false, b,
                                                   (short)0, c, false, false);
}

// 16x32 bf16 A-fragment from row-major bf16 LDS.
// lanes 0-15 (row=lane): K {+0..7, +16..23}; lanes 16-31: K {+8..15, +24..31}.
__device__ __forceinline__ v16bf load_afrag_lds(const char* base, int rowStride,
                                                int kByteOff, int lane) {
    int m  = lane & 15;
    int hi = lane >> 4;
    const char* p = base + m * rowStride + kByteOff + hi * 16;
    v16bf a;
    float4* o = reinterpret_cast<float4*>(&a);
    o[0] = *reinterpret_cast<const float4*>(p);
    o[1] = *reinterpret_cast<const float4*>(p + 32);
    return a;
}

// 32x16 bf16 B-fragment from pre-packed fragment-major weights:
// frag (nc,kc) is 1024 contiguous bytes; lane reads its 32 bytes -> 2x b128.
__device__ __forceinline__ v16bf load_bfrag_t(const char* __restrict__ Wt, int KC,
                                              int nc, int kc, int lane) {
    const char* p = Wt + (((size_t)nc * KC + kc) << 10) + lane * 32;
    v16bf b;
    float4* o = reinterpret_cast<float4*>(&b);
    o[0] = *reinterpret_cast<const float4*>(p);
    o[1] = *reinterpret_cast<const float4*>(p + 16);
    return b;
}

// Fallback: 32x16 B-fragment straight from row-major fp32 weights + cvt.
__device__ __forceinline__ v16bf load_bfrag_g(const float* __restrict__ W, int ldw,
                                              int kbase, int ncol, int lane) {
    int n  = lane & 15;
    int hi = lane >> 4;
    const float* p0 = W + (size_t)(kbase + hi * 8) * ldw + ncol + n;
    const float* p1 = p0 + (size_t)16 * ldw;
    v16bf b;
#pragma unroll
    for (int i = 0; i < 8; ++i) {
        b[i]     = (__bf16)p0[(size_t)i * ldw];
        b[i + 8] = (__bf16)p1[(size_t)i * ldw];
    }
    return b;
}

// ---------------------------------------------------------------------------
// Pack fp32 weights [E][Kdim][Nw] into WMMA B-fragment-major bf16 tiles:
// frag index = (e*NC + nc)*KC + kc, 1024 B each; word i of lane l holds
// K elements {base + (i>=4?16:0) + (l>>4)*8 + 2i%8, +1} of column nc*16+(l&15).
// ---------------------------------------------------------------------------
__global__ __launch_bounds__(256)
void pack_weights_kernel(const float* __restrict__ W, unsigned int* __restrict__ outw,
                         int KC, int NC, int Kdim, int Nw) {
    size_t idx = (size_t)blockIdx.x * 256 + threadIdx.x;   // one 32-bit word
    int    i   = (int)(idx & 7);
    int    l   = (int)((idx >> 3) & 31);
    size_t fr  = idx >> 8;
    int    kc  = (int)(fr % KC);
    size_t t2  = fr / KC;
    int    nc  = (int)(t2 % NC);
    int    e   = (int)(t2 / NC);
    const float* We = W + (size_t)e * Kdim * Nw;
    int hi = l >> 4;
    int k0 = kc * 32 + ((i >= 4) ? 16 : 0) + hi * 8 + ((2 * i) & 7);
    size_t col = (size_t)nc * 16 + (l & 15);
    __bf16 b0 = (__bf16)We[(size_t)k0 * Nw + col];
    __bf16 b1 = (__bf16)We[(size_t)(k0 + 1) * Nw + col];
    unsigned short u0, u1;
    __builtin_memcpy(&u0, &b0, 2);
    __builtin_memcpy(&u1, &b1, 2);
    outw[idx] = (unsigned)u0 | ((unsigned)u1 << 16);
}

// ---------------------------------------------------------------------------
// Router: logits = x @ Wr + br, softmax, top-2 (ties -> lowest index),
// normalized gates, per-expert compacted token lists.
// ---------------------------------------------------------------------------
__global__ __launch_bounds__(64)
void router_kernel(const float* __restrict__ x, const float* __restrict__ Wr,
                   const float* __restrict__ br, float* __restrict__ gates,
                   int* __restrict__ counts, int* __restrict__ tokList) {
    __shared__ float red[64][NEXP];
    int t   = blockIdx.x;
    int tid = threadIdx.x;

    float acc[NEXP];
#pragma unroll
    for (int e = 0; e < NEXP; ++e) acc[e] = 0.f;

    const float* xr = x + (size_t)t * DMODEL;
    for (int d = tid; d < DMODEL; d += 64) {
        float xv = xr[d];
        const float* wr = Wr + (size_t)d * NEXP;
#pragma unroll
        for (int e = 0; e < NEXP; ++e) acc[e] += xv * wr[e];
    }
#pragma unroll
    for (int e = 0; e < NEXP; ++e) red[tid][e] = acc[e];
    __syncthreads();

    if (tid == 0) {
        float lg[NEXP];
#pragma unroll
        for (int e = 0; e < NEXP; ++e) {
            float s = 0.f;
            for (int i = 0; i < 64; ++i) s += red[i][e];
            lg[e] = s + br[e];
        }
        float mx = lg[0];
#pragma unroll
        for (int e = 1; e < NEXP; ++e) mx = fmaxf(mx, lg[e]);
        float p[NEXP], se = 0.f;
#pragma unroll
        for (int e = 0; e < NEXP; ++e) { p[e] = expf(lg[e] - mx); se += p[e]; }
        float inv = 1.f / se;
#pragma unroll
        for (int e = 0; e < NEXP; ++e) p[e] *= inv;

        int i1 = 0;
#pragma unroll
        for (int e = 1; e < NEXP; ++e) if (p[e] > p[i1]) i1 = e;
        int i2 = (i1 == 0) ? 1 : 0;
#pragma unroll
        for (int e = 0; e < NEXP; ++e)
            if (e != i1 && p[e] > p[i2]) i2 = e;

        float g1 = p[i1], g2 = p[i2];
        float wsum = g1 + g2;
        g1 /= wsum; g2 /= wsum;

        float go[NEXP];
#pragma unroll
        for (int e = 0; e < NEXP; ++e) go[e] = 0.f;
        go[i1] = g1; go[i2] = g2;
#pragma unroll
        for (int e = 0; e < NEXP; ++e) gates[(size_t)t * NEXP + e] = go[e];

        int s1 = atomicAdd(&counts[i1], 1); tokList[(size_t)i1 * N_TOK + s1] = t;
        int s2 = atomicAdd(&counts[i2], 1); tokList[(size_t)i2 * N_TOK + s2] = t;
    }
}

// ---------------------------------------------------------------------------
// Fused expert FFN, 32-token tiles (2 M-subtiles), hidden in LDS slabs.
// out[t] += gate * ( gelu(Xg @ W1[e] + b1[e]) @ W2[e] + b2[e] )
// ---------------------------------------------------------------------------
template <bool TILED>
__global__ __launch_bounds__(256, 1)
void expert_ffn_kernel(const float* __restrict__ x,
                       const float* __restrict__ W1, const float* __restrict__ b1,
                       const float* __restrict__ W2, const float* __restrict__ b2,
                       const char* __restrict__ W1t, const char* __restrict__ W2t,
                       const float* __restrict__ gates,
                       const int* __restrict__ counts,
                       const int* __restrict__ tokList,
                       float* __restrict__ out) {
    extern __shared__ char smem[];
    char* xlds = smem;                        // MTILE * XSTR (bf16 X tile)
    char* hlds = smem + MTILE * XSTR;         // MTILE * HSTR (bf16 hidden slab)
    __shared__ int   sTok[MTILE];
    __shared__ float sGate[MTILE];

    const int e    = blockIdx.x / TILES_M;
    const int tile = blockIdx.x % TILES_M;
    const int cnt  = counts[e];
    const int m0   = tile * MTILE;
    if (m0 >= cnt) return;                    // uniform early-exit

    const int tid  = threadIdx.x;
    const int lane = tid & 31;
    const int wv   = tid >> 5;                // 8 waves

    if (tid < MTILE) {
        int gm = m0 + tid;
        if (gm < cnt) {
            int t = tokList[(size_t)e * N_TOK + gm];
            sTok[tid]  = t;
            sGate[tid] = gates[(size_t)t * NEXP + e];
        } else {
            sTok[tid]  = -1;
            sGate[tid] = 0.f;
        }
    }
    __syncthreads();

    // Stage gathered X tile (fp32 -> bf16) row-major into LDS.
    for (int idx = tid; idx < MTILE * DMODEL; idx += 256) {
        int r = idx >> 10;                    // / DMODEL
        int c = idx & (DMODEL - 1);
        int t = sTok[r];
        float v = (t >= 0) ? x[(size_t)t * DMODEL + c] : 0.f;
        *reinterpret_cast<__bf16*>(xlds + r * XSTR + c * 2) = (__bf16)v;
    }
    __syncthreads();

    const float* W1e  = W1 + (size_t)e * DMODEL * HDIM;
    const float* b1e  = b1 + (size_t)e * HDIM;
    const float* W2e  = W2 + (size_t)e * HDIM * DMODEL;
    const float* b2e  = b2 + (size_t)e * DMODEL;
    const char*  W1te = W1t + (size_t)e * W1T_BYTES_PER_E;
    const char*  W2te = W2t + (size_t)e * W2T_BYTES_PER_E;

    const int nloc = lane & 15;
    const int mhi  = (lane >> 4) * 8;

    const v8f vzero = (v8f){0.f,0.f,0.f,0.f,0.f,0.f,0.f,0.f};
    v8f acc2a[8], acc2b[8];                   // GEMM2 accumulators, both M-subtiles
#pragma unroll
    for (int j = 0; j < 8; ++j) { acc2a[j] = vzero; acc2b[j] = vzero; }

    for (int s = 0; s < HDIM / SLAB; ++s) {   // 8 slabs of 512 hidden cols
        // ---- GEMM1 into this slab: 2 passes of 2 n-chunks per wave ----
        // kc-inner shares each A-fragment pair across 2 B fragments (4 WMMA/kc).
#pragma unroll 1
        for (int jp = 0; jp < 2; ++jp) {
            int lc0 = wv * 4 + jp * 2;        // slab-local n-chunk (0..31)
            int lc1 = lc0 + 1;
            int nc0 = s * 32 + lc0;           // global n-chunk (0..255)
            int nc1 = nc0 + 1;
            v8f c00 = vzero, c01 = vzero, c10 = vzero, c11 = vzero;
#pragma unroll 1
            for (int kc = 0; kc < KC1; ++kc) {
                v16bf a0 = load_afrag_lds(xlds,             XSTR, kc * 64, lane);
                v16bf a1 = load_afrag_lds(xlds + 16 * XSTR, XSTR, kc * 64, lane);
                v16bf b0 = TILED ? load_bfrag_t(W1te, KC1, nc0, kc, lane)
                                 : load_bfrag_g(W1e, HDIM, kc * 32, nc0 * 16, lane);
                v16bf b1 = TILED ? load_bfrag_t(W1te, KC1, nc1, kc, lane)
                                 : load_bfrag_g(W1e, HDIM, kc * 32, nc1 * 16, lane);
                c00 = wmma_bf16(a0, b0, c00);
                c01 = wmma_bf16(a1, b0, c01);
                c10 = wmma_bf16(a0, b1, c10);
                c11 = wmma_bf16(a1, b1, c11);
            }
            // bias + exact gelu + store both chunks to hidden slab (bf16)
            float bb0 = b1e[nc0 * 16 + nloc];
            float bb1 = b1e[nc1 * 16 + nloc];
            char* h00 = hlds + mhi * HSTR + (lc0 * 16 + nloc) * 2;
            char* h10 = hlds + mhi * HSTR + (lc1 * 16 + nloc) * 2;
#pragma unroll
            for (int r = 0; r < 8; ++r) {
                *reinterpret_cast<__bf16*>(h00 + r * HSTR)             = (__bf16)gelu_exact(c00[r] + bb0);
                *reinterpret_cast<__bf16*>(h00 + (16 + r - 0) * HSTR + 16 * HSTR - 16 * HSTR) ;
                *reinterpret_cast<__bf16*>(h00 + r * HSTR + 16 * HSTR) = (__bf16)gelu_exact(c01[r] + bb0);
                *reinterpret_cast<__bf16*>(h10 + r * HSTR)             = (__bf16)gelu_exact(c10[r] + bb1);
                *reinterpret_cast<__bf16*>(h10 + r * HSTR + 16 * HSTR) = (__bf16)gelu_exact(c11[r] + bb1);
            }
        }
        __syncthreads();                      // hidden slab visible

        // ---- GEMM2 partial: kc-outer, A loaded once, 8 n-chunks inner ----
#pragma unroll 1
        for (int kc = 0; kc < SLAB / 32; ++kc) {
            v16bf a0 = load_afrag_lds(hlds,             HSTR, kc * 64, lane);
            v16bf a1 = load_afrag_lds(hlds + 16 * HSTR, HSTR, kc * 64, lane);
            int kcg = s * 16 + kc;            // global k-chunk for packed W2
#pragma unroll
            for (int j = 0; j < 8; ++j) {     // fully unrolled: acc stays in VGPRs
                int nc2 = wv * 8 + j;
                v16bf b = TILED ? load_bfrag_t(W2te, KC2, nc2, kcg, lane)
                                : load_bfrag_g(W2e, DMODEL, s * SLAB + kc * 32,
                                               nc2 * 16, lane);
                acc2a[j] = wmma_bf16(a0, b, acc2a[j]);
                acc2b[j] = wmma_bf16(a1, b, acc2b[j]);
            }
        }
        __syncthreads();                      // before slab is overwritten
    }

    // ---- Epilogue: gate-scaled accumulate into output ----
#pragma unroll
    for (int j = 0; j < 8; ++j) {
        int n2 = (wv * 8 + j) * 16;
        float bb = b2e[n2 + nloc];
#pragma unroll
        for (int r = 0; r < 8; ++r) {
            int m  = mhi + r;
            int t0 = sTok[m];
            if (t0 >= 0)
                atomicAdd(&out[(size_t)t0 * DMODEL + n2 + nloc],
                          sGate[m] * (acc2a[j][r] + bb));
            int t1 = sTok[16 + m];
            if (t1 >= 0)
                atomicAdd(&out[(size_t)t1 * DMODEL + n2 + nloc],
                          sGate[16 + m] * (acc2b[j][r] + bb));
        }
    }
}

// ---------------------------------------------------------------------------
extern "C" void kernel_launch(void* const* d_in, const int* in_sizes, int n_in,
                              void* d_out, int out_size, void* d_ws, size_t ws_size,
                              hipStream_t stream) {
    (void)in_sizes; (void)n_in;
    const float* x  = (const float*)d_in[0];
    const float* Wr = (const float*)d_in[1];
    const float* br = (const float*)d_in[2];
    const float* W1 = (const float*)d_in[3];
    const float* b1 = (const float*)d_in[4];
    const float* W2 = (const float*)d_in[5];
    const float* b2 = (const float*)d_in[6];
    float* out = (float*)d_out;

    char* ws      = (char*)d_ws;
    int*  counts  = (int*)ws;                                      // 8 ints
    int*  tokList = (int*)(ws + 256);                              // E*N ints
    float* gates  = (float*)(ws + 256 + (size_t)NEXP * N_TOK * sizeof(int));

    const size_t offW1t = (size_t)1 << 20;                         // 1 MB
    const size_t offW2t = offW1t + NEXP * W1T_BYTES_PER_E;         // +64 MB
    const size_t wsReq  = offW2t + NEXP * W2T_BYTES_PER_E;         // ~135 MB
    char* w1t = ws + offW1t;
    char* w2t = ws + offW2t;
    const bool tiled = (ws_size >= wsReq);

    hipMemsetAsync(counts, 0, NEXP * sizeof(int), stream);
    hipMemsetAsync(out, 0, (size_t)out_size * sizeof(float), stream);

    if (tiled) {
        unsigned g1 = (unsigned)((size_t)NEXP * KC1 * NC1 * 256 / 256);
        unsigned g2 = (unsigned)((size_t)NEXP * KC2 * NC2 * 256 / 256);
        pack_weights_kernel<<<g1, 256, 0, stream>>>(W1, (unsigned int*)w1t,
                                                    KC1, NC1, DMODEL, HDIM);
        pack_weights_kernel<<<g2, 256, 0, stream>>>(W2, (unsigned int*)w2t,
                                                    KC2, NC2, HDIM, DMODEL);
    }

    router_kernel<<<N_TOK, 64, 0, stream>>>(x, Wr, br, gates, counts, tokList);

    size_t smem = (size_t)MTILE * XSTR + (size_t)MTILE * HSTR;     // ~99 KB
    if (tiled) {
        expert_ffn_kernel<true><<<NEXP * TILES_M, 256, smem, stream>>>(
            x, W1, b1, W2, b2, w1t, w2t, gates, counts, tokList, out);
    } else {
        expert_ffn_kernel<false><<<NEXP * TILES_M, 256, smem, stream>>>(
            x, W1, b1, W2, b2, w1t, w2t, gates, counts, tokList, out);
    }
}
